// marginalBert_25881472926362
// MI455X (gfx1250) — matrix operand order repaired and checked by reading, not compiled, not run on hardware
//
#include <hip/hip_runtime.h>
#include <hip/hip_bf16.h>
#include <math.h>

// ---------------- model constants (match reference) ----------------
#define VOCABSZ 30522
#define H       768
#define NLAYERS 12
#define HEADS   12
#define DHD     64
#define FFD     3072
#define SEQ     64
#define BATCH   128
#define TOK     (BATCH * SEQ)
#define LN_EPS  1e-12f
#define MARGIN  0.1f
#define NBEST   8
#define NGROUP  16

enum { ACT_NONE = 0, ACT_GELU = 1, ACT_TANH = 2 };

typedef _Float16 half8  __attribute__((ext_vector_type(8)));
typedef _Float16 half16 __attribute__((ext_vector_type(16)));
typedef float    float8 __attribute__((ext_vector_type(8)));

// ---------------- helpers ----------------
__device__ __forceinline__ float block_reduce_sum(float v, float* sbuf) {
    const int t = threadIdx.x;
    sbuf[t] = v;
    __syncthreads();
    for (int s = blockDim.x >> 1; s > 0; s >>= 1) {
        if (t < s) sbuf[t] += sbuf[t + s];
        __syncthreads();
    }
    float r = sbuf[0];
    __syncthreads();
    return r;
}

// ---------------- embedding + layernorm ----------------
// one block per token row; 256 threads, 3 columns each
__global__ void __launch_bounds__(256) embed_ln_kernel(
    const int* __restrict__ ids, const float* __restrict__ wemb,
    const float* __restrict__ pemb, const float* __restrict__ temb,
    const float* __restrict__ g, const float* __restrict__ b,
    float* __restrict__ xf, _Float16* __restrict__ xh)
{
    __shared__ float red[256];
    const int row = blockIdx.x;
    const int s   = row % SEQ;
    const int id  = ids[row];
    float vals[3];
    float lsum = 0.f;
#pragma unroll
    for (int i = 0; i < 3; ++i) {
        const int c = threadIdx.x + i * 256;
        float v = wemb[(size_t)id * H + c] + pemb[(size_t)s * H + c] + temb[c];
        vals[i] = v;
        lsum += v;
    }
    const float mean = block_reduce_sum(lsum, red) * (1.0f / H);
    float lsq = 0.f;
#pragma unroll
    for (int i = 0; i < 3; ++i) { float d = vals[i] - mean; lsq += d * d; }
    const float var  = block_reduce_sum(lsq, red) * (1.0f / H);
    const float rstd = rsqrtf(var + LN_EPS);
#pragma unroll
    for (int i = 0; i < 3; ++i) {
        const int c = threadIdx.x + i * 256;
        const float o = (vals[i] - mean) * rstd * g[c] + b[c];
        xf[(size_t)row * H + c] = o;
        xh[(size_t)row * H + c] = (_Float16)o;
    }
}

// ---------------- residual add + layernorm (in-place safe per row) ----------------
__global__ void __launch_bounds__(256) add_ln_kernel(
    const float* __restrict__ res, const float* __restrict__ delta,
    const float* __restrict__ g, const float* __restrict__ b,
    float* __restrict__ outf, _Float16* __restrict__ outh)
{
    __shared__ float red[256];
    const int row = blockIdx.x;
    float vals[3];
    float lsum = 0.f;
#pragma unroll
    for (int i = 0; i < 3; ++i) {
        const int c = threadIdx.x + i * 256;
        float v = res[(size_t)row * H + c] + delta[(size_t)row * H + c];
        vals[i] = v;
        lsum += v;
    }
    const float mean = block_reduce_sum(lsum, red) * (1.0f / H);
    float lsq = 0.f;
#pragma unroll
    for (int i = 0; i < 3; ++i) { float d = vals[i] - mean; lsq += d * d; }
    const float var  = block_reduce_sum(lsq, red) * (1.0f / H);
    const float rstd = rsqrtf(var + LN_EPS);
#pragma unroll
    for (int i = 0; i < 3; ++i) {
        const int c = threadIdx.x + i * 256;
        const float o = (vals[i] - mean) * rstd * g[c] + b[c];
        outf[(size_t)row * H + c] = o;
        outh[(size_t)row * H + c] = (_Float16)o;
    }
}

// ---------------- LDS-tiled transpose + f32->f16 convert: W[K,N] -> Wt[N,K] ----------------
__global__ void __launch_bounds__(256) transpose_f32_to_f16(
    const float* __restrict__ W, _Float16* __restrict__ Wt, int K, int N)
{
    __shared__ float tile[32][33];
    const int nb = blockIdx.x * 32;
    const int kb = blockIdx.y * 32;
    const int tx = threadIdx.x;   // 0..31
    const int ty = threadIdx.y;   // 0..7
#pragma unroll
    for (int i = 0; i < 32; i += 8)
        tile[ty + i][tx] = W[(size_t)(kb + ty + i) * N + nb + tx];
    __syncthreads();
#pragma unroll
    for (int i = 0; i < 32; i += 8)
        Wt[(size_t)(nb + ty + i) * K + kb + tx] = (_Float16)tile[tx][ty + i];
}

// ---------------- WMMA GEMM: C[M,N] = A[M,K](f16,row) * Wt[N,K](f16,N-major) + bias ----------------
// grid = (N/128, M/64), block = 128 threads (4 waves).
// Wave tile 32x64 = 2(M) x 4(N) wmma_f32_16x16x32_f16 accumulators:
// 12 b128 fragment loads per K=32 step feed 8 WMMAs -> ~21.3 FLOP/byte from L2.
// K loop unrolled 2x so the next step's loads issue under the current WMMAs.
__device__ __forceinline__ void store_tile(
    float8 acc, int mb, int nb, int lane, const float* __restrict__ bias,
    float* __restrict__ Cf, _Float16* __restrict__ Ch, int N, int act)
{
    const int half_ = lane >> 4;
    const int lmod  = lane & 15;
    const int col   = nb + lmod;
    const float bv  = bias ? bias[col] : 0.f;
#pragma unroll
    for (int r = 0; r < 8; ++r) {
        const int row = mb + r + 8 * half_;
        float v = acc[r] + bv;
        if (act == ACT_GELU)      v = 0.5f * v * (1.0f + erff(v * 0.70710678118654752f));
        else if (act == ACT_TANH) v = tanhf(v);
        if (Cf) Cf[(size_t)row * N + col] = v;
        if (Ch) Ch[(size_t)row * N + col] = (_Float16)v;
    }
}

__global__ void __launch_bounds__(128) gemm_wmma_kernel(
    const _Float16* __restrict__ A, const _Float16* __restrict__ Wt,
    const float* __restrict__ bias, float* __restrict__ Cf,
    _Float16* __restrict__ Ch, int M, int N, int K, int act)
{
    const int lane  = threadIdx.x & 31;
    const int wave  = threadIdx.x >> 5;           // 0..3
    const int half_ = lane >> 4;                  // lane group 0-15 / 16-31
    const int lmod  = lane & 15;
    const int m0 = blockIdx.y * 64  + (wave >> 1) * 32;
    const int n0 = blockIdx.x * 128 + (wave & 1) * 64;

    float8 acc[2][4];
#pragma unroll
    for (int i = 0; i < 2; ++i)
#pragma unroll
        for (int j = 0; j < 4; ++j) acc[i][j] = (float8){};

    // A fragment rows (16x32 f16 layout): lane -> M = m + lmod, K halves by lane group
    const _Float16* ap[2];
#pragma unroll
    for (int i = 0; i < 2; ++i)
        ap[i] = A + (size_t)(m0 + 16 * i + lmod) * K + half_ * 16;
    // B fragment cols (32x16 f16 layout, Wt is N-major so rows are contiguous K)
    const _Float16* bp[4];
#pragma unroll
    for (int j = 0; j < 4; ++j)
        bp[j] = Wt + (size_t)(n0 + 16 * j + lmod) * K + half_ * 16;

#pragma unroll 2
    for (int kk = 0; kk < K; kk += 32) {
        half16 afrag[2], bfrag[4];
#pragma unroll
        for (int i = 0; i < 2; ++i) {
            half8 lo = *(const half8*)(ap[i] + kk);
            half8 hi = *(const half8*)(ap[i] + kk + 8);
            afrag[i] = __builtin_shufflevector(lo, hi, 0,1,2,3,4,5,6,7,8,9,10,11,12,13,14,15);
        }
#pragma unroll
        for (int j = 0; j < 4; ++j) {
            half8 lo = *(const half8*)(bp[j] + kk);
            half8 hi = *(const half8*)(bp[j] + kk + 8);
            bfrag[j] = __builtin_shufflevector(lo, hi, 0,1,2,3,4,5,6,7,8,9,10,11,12,13,14,15);
        }
#pragma unroll
        for (int i = 0; i < 2; ++i)
#pragma unroll
            for (int j = 0; j < 4; ++j)
                acc[i][j] = __builtin_amdgcn_wmma_f32_16x16x32_f16(
                    false, afrag[i], false, bfrag[j], (short)0, acc[i][j], false, false);
    }

#pragma unroll
    for (int i = 0; i < 2; ++i)
#pragma unroll
        for (int j = 0; j < 4; ++j)
            store_tile(acc[i][j], m0 + 16 * i, n0 + 16 * j, lane, bias, Cf, Ch, N, act);
}

// ---------------- attention: one block per (batch, head); 64 threads = 64 query rows ----------------
__global__ void __launch_bounds__(64) attention_kernel(
    const _Float16* __restrict__ Q, const _Float16* __restrict__ Kh,
    const _Float16* __restrict__ V, const int* __restrict__ mask,
    _Float16* __restrict__ ctx)
{
    const int bh = blockIdx.x;
    const int b  = bh / HEADS;
    const int h  = bh % HEADS;
    const int q  = threadIdx.x;   // 0..63

    __shared__ _Float16 Ks[SEQ][DHD + 8];
    __shared__ _Float16 Vs[SEQ][DHD + 8];
    __shared__ float    Ss[SEQ][SEQ + 1];

    const size_t base = ((size_t)(b * SEQ + q)) * H + h * DHD;
    const _Float16* kp = Kh + base;
    const _Float16* vp = V + base;
#pragma unroll 8
    for (int d = 0; d < DHD; ++d) { Ks[q][d] = kp[d]; Vs[q][d] = vp[d]; }

    float qr[DHD];
    const _Float16* qp = Q + base;
#pragma unroll 8
    for (int d = 0; d < DHD; ++d) qr[d] = (float)qp[d];
    __syncthreads();

    float mx = -1e30f;
    for (int k = 0; k < SEQ; ++k) {
        float s = 0.f;
#pragma unroll 8
        for (int d = 0; d < DHD; ++d) s += qr[d] * (float)Ks[k][d];
        s = s * 0.125f + (1.0f - (float)mask[b * SEQ + k]) * (-10000.0f);
        Ss[q][k] = s;
        mx = fmaxf(mx, s);
    }
    float sum = 0.f;
    for (int k = 0; k < SEQ; ++k) {
        float e = __expf(Ss[q][k] - mx);
        Ss[q][k] = e;
        sum += e;
    }
    const float inv = 1.0f / sum;

    float out[DHD];
#pragma unroll 8
    for (int d = 0; d < DHD; ++d) out[d] = 0.f;
    for (int k = 0; k < SEQ; ++k) {
        const float p = Ss[q][k] * inv;
#pragma unroll 8
        for (int d = 0; d < DHD; ++d) out[d] += p * (float)Vs[k][d];
    }
    _Float16* op = ctx + base;
#pragma unroll 8
    for (int d = 0; d < DHD; ++d) op[d] = (_Float16)out[d];
}

// ---------------- gather [CLS] rows into compact f16 matrix ----------------
__global__ void __launch_bounds__(256) gather_cls_kernel(
    const _Float16* __restrict__ xh, _Float16* __restrict__ clsh)
{
    const int b = blockIdx.x;
#pragma unroll
    for (int i = 0; i < 3; ++i) {
        const int c = threadIdx.x + i * 256;
        clsh[(size_t)b * H + c] = xh[(size_t)b * SEQ * H + c];
    }
}

// ---------------- final: linear head, per-group softmax, CE + margin loss ----------------
__global__ void __launch_bounds__(128) final_kernel(
    const float* __restrict__ pooled, const float* __restrict__ am,
    const float* __restrict__ ctc, const float* __restrict__ Wl,
    const float* __restrict__ bl, const float* __restrict__ labels,
    const int* __restrict__ wer_rank, float* __restrict__ out)
{
    __shared__ float ss[BATCH];
    __shared__ float gloss[NGROUP];
    const int tid = threadIdx.x;   // 0..127

    float s = bl[0];
    const float* pr = pooled + (size_t)tid * H;
    for (int j = 0; j < H; ++j) s += pr[j] * Wl[j];
    s += am[tid] * Wl[H] + ctc[tid] * Wl[H + 1];
    out[tid] = s;
    ss[tid]  = s;
    __syncthreads();

    if (tid < NGROUP) {
        const int g = tid;
        float mx = -1e30f;
        for (int i = 0; i < NBEST; ++i) mx = fmaxf(mx, ss[g * NBEST + i]);
        float e[NBEST], sum = 0.f;
        for (int i = 0; i < NBEST; ++i) { e[i] = __expf(ss[g * NBEST + i] - mx); sum += e[i]; }
        float p[NBEST];
        float l = 0.f;
        for (int i = 0; i < NBEST; ++i) {
            p[i] = e[i] / sum;
            if (labels[g * NBEST + i] > 0.5f) l += -logf(p[i]);
        }
        for (int i = 0; i < NBEST; ++i) {
            for (int j = i + 1; j < NBEST; ++j) {
                const int ri = wer_rank[g * NBEST + i];
                const int rj = wer_rank[g * NBEST + j];
                const float d = p[rj] - p[ri] + MARGIN;
                if (d > 0.f) l += d;
            }
        }
        gloss[g] = l;
    }
    __syncthreads();
    if (tid == 0) {
        float t = 0.f;
        for (int g = 0; g < NGROUP; ++g) t += gloss[g];
        out[BATCH] = t / (float)BATCH;
    }
}

// ---------------- host: orchestrate full forward ----------------
// input order (setup_inputs dict, params flattened in insertion order):
enum {
    IN_IDS = 0, IN_MASK, IN_AM, IN_CTC, IN_WER, IN_LABELS, IN_NBEST,
    IN_WORD_EMB, IN_POS_EMB, IN_TYPE_EMB, IN_EMB_LN_G, IN_EMB_LN_B,
    IN_WQ, IN_BQ, IN_WK, IN_BK, IN_WV, IN_BV, IN_WO, IN_BO,
    IN_LN1_G, IN_LN1_B, IN_W1, IN_B1, IN_W2, IN_B2, IN_LN2_G, IN_LN2_B,
    IN_WP, IN_BP, IN_WL, IN_BL
};

extern "C" void kernel_launch(void* const* d_in, const int* in_sizes, int n_in,
                              void* d_out, int out_size, void* d_ws, size_t ws_size,
                              hipStream_t stream) {
    (void)in_sizes; (void)n_in; (void)out_size; (void)ws_size;

    const int*   ids    = (const int*)d_in[IN_IDS];
    const int*   mask   = (const int*)d_in[IN_MASK];
    const float* am     = (const float*)d_in[IN_AM];
    const float* ctc    = (const float*)d_in[IN_CTC];
    const int*   wer    = (const int*)d_in[IN_WER];
    const float* labels = (const float*)d_in[IN_LABELS];
    const float* wemb   = (const float*)d_in[IN_WORD_EMB];
    const float* pemb   = (const float*)d_in[IN_POS_EMB];
    const float* temb   = (const float*)d_in[IN_TYPE_EMB];
    const float* elng   = (const float*)d_in[IN_EMB_LN_G];
    const float* elnb   = (const float*)d_in[IN_EMB_LN_B];
    const float* Wq = (const float*)d_in[IN_WQ]; const float* bq = (const float*)d_in[IN_BQ];
    const float* Wk = (const float*)d_in[IN_WK]; const float* bk = (const float*)d_in[IN_BK];
    const float* Wv = (const float*)d_in[IN_WV]; const float* bv = (const float*)d_in[IN_BV];
    const float* Wo = (const float*)d_in[IN_WO]; const float* bo = (const float*)d_in[IN_BO];
    const float* g1 = (const float*)d_in[IN_LN1_G]; const float* b1n = (const float*)d_in[IN_LN1_B];
    const float* W1 = (const float*)d_in[IN_W1]; const float* fb1 = (const float*)d_in[IN_B1];
    const float* W2 = (const float*)d_in[IN_W2]; const float* fb2 = (const float*)d_in[IN_B2];
    const float* g2 = (const float*)d_in[IN_LN2_G]; const float* b2n = (const float*)d_in[IN_LN2_B];
    const float* Wp = (const float*)d_in[IN_WP]; const float* bp = (const float*)d_in[IN_BP];
    const float* Wl = (const float*)d_in[IN_WL]; const float* bl = (const float*)d_in[IN_BL];

    // workspace carve (~180 MB)
    char* p = (char*)d_ws;
    auto carve = [&](size_t bytes) -> void* {
        void* r = (void*)p;
        p += (bytes + 255) & ~(size_t)255;
        return r;
    };
    float*     xf   = (float*)    carve((size_t)TOK * H * 4);
    _Float16*  xh   = (_Float16*) carve((size_t)TOK * H * 2);
    _Float16*  qh   = (_Float16*) carve((size_t)TOK * H * 2);
    _Float16*  kh   = (_Float16*) carve((size_t)TOK * H * 2);
    _Float16*  vh   = (_Float16*) carve((size_t)TOK * H * 2);
    _Float16*  ctxh = (_Float16*) carve((size_t)TOK * H * 2);
    float*     tmpf = (float*)    carve((size_t)TOK * H * 4);
    _Float16*  ffh  = (_Float16*) carve((size_t)TOK * FFD * 2);
    _Float16*  wqT  = (_Float16*) carve((size_t)H * H * 2);
    _Float16*  wkT  = (_Float16*) carve((size_t)H * H * 2);
    _Float16*  wvT  = (_Float16*) carve((size_t)H * H * 2);
    _Float16*  woT  = (_Float16*) carve((size_t)H * H * 2);
    _Float16*  w1T  = (_Float16*) carve((size_t)H * FFD * 2);
    _Float16*  w2T  = (_Float16*) carve((size_t)FFD * H * 2);
    _Float16*  wpT  = (_Float16*) carve((size_t)H * H * 2);
    _Float16*  clsh = (_Float16*) carve((size_t)BATCH * H * 2);
    float*     poolf= (float*)    carve((size_t)BATCH * H * 4);

    const dim3 trBlk(32, 8);
    const dim3 trHH(H / 32, H / 32);        // [H,H] weights
    const dim3 trHF(FFD / 32, H / 32);      // W1: [H,FF]
    const dim3 trFH(H / 32, FFD / 32);      // W2: [FF,H]
    const dim3 gemmHH(H / 128, TOK / 64);   // M=TOK, N=H
    const dim3 gemmHF(FFD / 128, TOK / 64); // M=TOK, N=FF

    // embeddings + LN
    embed_ln_kernel<<<TOK, 256, 0, stream>>>(ids, wemb, pemb, temb, elng, elnb, xf, xh);

    for (int l = 0; l < NLAYERS; ++l) {
        const size_t oHH = (size_t)l * H * H;
        const size_t oHF = (size_t)l * H * FFD;
        const size_t oH  = (size_t)l * H;
        const size_t oF  = (size_t)l * FFD;

        transpose_f32_to_f16<<<trHH, trBlk, 0, stream>>>(Wq + oHH, wqT, H, H);
        transpose_f32_to_f16<<<trHH, trBlk, 0, stream>>>(Wk + oHH, wkT, H, H);
        transpose_f32_to_f16<<<trHH, trBlk, 0, stream>>>(Wv + oHH, wvT, H, H);
        transpose_f32_to_f16<<<trHH, trBlk, 0, stream>>>(Wo + oHH, woT, H, H);
        transpose_f32_to_f16<<<trHF, trBlk, 0, stream>>>(W1 + oHF, w1T, H, FFD);
        transpose_f32_to_f16<<<trFH, trBlk, 0, stream>>>(W2 + oHF, w2T, FFD, H);

        // Q, K, V projections (f16 outputs for attention)
        gemm_wmma_kernel<<<gemmHH, 128, 0, stream>>>(xh, wqT, bq + oH, nullptr, qh, TOK, H, H, ACT_NONE);
        gemm_wmma_kernel<<<gemmHH, 128, 0, stream>>>(xh, wkT, bk + oH, nullptr, kh, TOK, H, H, ACT_NONE);
        gemm_wmma_kernel<<<gemmHH, 128, 0, stream>>>(xh, wvT, bv + oH, nullptr, vh, TOK, H, H, ACT_NONE);

        attention_kernel<<<BATCH * HEADS, 64, 0, stream>>>(qh, kh, vh, mask, ctxh);

        // output projection + residual LN
        gemm_wmma_kernel<<<gemmHH, 128, 0, stream>>>(ctxh, woT, bo + oH, tmpf, nullptr, TOK, H, H, ACT_NONE);
        add_ln_kernel<<<TOK, 256, 0, stream>>>(xf, tmpf, g1 + oH, b1n + oH, xf, xh);

        // FFN
        gemm_wmma_kernel<<<gemmHF, 128, 0, stream>>>(xh, w1T, fb1 + oF, nullptr, ffh, TOK, FFD, H, ACT_GELU);
        gemm_wmma_kernel<<<gemmHH, 128, 0, stream>>>(ffh, w2T, fb2 + oH, tmpf, nullptr, TOK, H, FFD, ACT_NONE);
        add_ln_kernel<<<TOK, 256, 0, stream>>>(xf, tmpf, g2 + oH, b2n + oH, xf, xh);
    }

    // pooler: tanh(cls @ Wp + bp)
    gather_cls_kernel<<<BATCH, 256, 0, stream>>>(xh, clsh);
    transpose_f32_to_f16<<<trHH, trBlk, 0, stream>>>(Wp, wpT, H, H);
    gemm_wmma_kernel<<<dim3(H / 128, BATCH / 64), 128, 0, stream>>>(
        clsh, wpT, bp, poolf, nullptr, BATCH, H, H, ACT_TANH);

    // linear head + softmax-over-nbest + CE + margin loss
    final_kernel<<<1, 128, 0, stream>>>(poolf, am, ctc, Wl, bl, labels, wer, (float*)d_out);
}
